// SLPFilteredNormalized_30528627540483
// MI455X (gfx1250) — matrix operand order repaired and checked by reading, not compiled, not run on hardware
//
#include <hip/hip_runtime.h>
#include <hip/hip_bf16.h>

// MI455X / gfx1250, wave32. WMMA f32_16x16x32_bf16 path.
// B=8192, N=4096, M=256.
typedef __attribute__((ext_vector_type(16))) __bf16 v16bf;
typedef __attribute__((ext_vector_type(8)))  float  v8f;

#define BB 8192
#define NN 4096
#define MM 256

__device__ __forceinline__ unsigned short f2bf(float f) {
    unsigned u = __builtin_bit_cast(unsigned, f);
    unsigned r = (u + 0x7FFFu + ((u >> 16) & 1u)) >> 16;   // RNE
    return (unsigned short)r;
}

union FragU { v16bf v; int4 q[2]; };

// ---------------------------------------------------------------------------
// Kernel 1: H = relu(x @ W1 + b1)  -> bf16 [B, 256] in ws
//           mask[b, n] = (x[b,n] != 0) packed 16 bits/ushort -> ws
// grid = B/64 blocks, 256 threads (8 waves; wave grid 4 rows x 2 cols;
// each wave owns 16 rows x 128 cols = 8 WMMA accum tiles).
// ---------------------------------------------------------------------------
__launch_bounds__(256)
__global__ void k1_gemm1_relu_mask(const float* __restrict__ x,
                                   const float* __restrict__ W1,
                                   const float* __restrict__ b1,
                                   unsigned short* __restrict__ Hout,
                                   unsigned short* __restrict__ maskOut) {
    __shared__ unsigned short xT[64][72];    // [row][k]  row stride 144B (16B aligned)
    __shared__ unsigned short w1T[256][72];  // [col][k]  col-major, stride 144B
    __shared__ float b1s[256];

    const int tid  = threadIdx.x;
    const int lane = tid & 31;
    const int wave = tid >> 5;
    const int waveRow = wave >> 1;          // 0..3 -> 16 rows each
    const int waveCol = wave & 1;           // 0..1 -> 128 cols each
    const int rowW = waveRow << 4;
    const int colW = waveCol << 7;
    const int rowBase = blockIdx.x * 64;

    b1s[tid] = b1[tid];

    v8f acc[8];
#pragma unroll
    for (int t = 0; t < 8; ++t) acc[t] = (v8f)0.0f;

    const int xr  = tid >> 2;               // 0..63 local row
    const int xc4 = (tid & 3) << 4;         // 0,16,32,48 local k base
    const int w1c = tid;                    // 0..255: this thread owns column w1c

    for (int kk = 0; kk < NN; kk += 64) {
        // ---- stage x tile (fp32 -> bf16) + emit packed mask -------------
        {
            const float4* xp =
                (const float4*)(x + (size_t)(rowBase + xr) * NN + kk + xc4);
            unsigned short tmpx[16];
            unsigned m16 = 0;
#pragma unroll
            for (int j = 0; j < 4; ++j) {
                float4 f = xp[j];
                tmpx[j * 4 + 0] = f2bf(f.x);
                tmpx[j * 4 + 1] = f2bf(f.y);
                tmpx[j * 4 + 2] = f2bf(f.z);
                tmpx[j * 4 + 3] = f2bf(f.w);
                m16 |= (f.x != 0.0f ? 1u : 0u) << (j * 4 + 0);
                m16 |= (f.y != 0.0f ? 1u : 0u) << (j * 4 + 1);
                m16 |= (f.z != 0.0f ? 1u : 0u) << (j * 4 + 2);
                m16 |= (f.w != 0.0f ? 1u : 0u) << (j * 4 + 3);
            }
            int4* dst = (int4*)&xT[xr][xc4];
            dst[0] = ((const int4*)tmpx)[0];
            dst[1] = ((const int4*)tmpx)[1];
            maskOut[(size_t)(rowBase + xr) * (NN / 16) + ((kk + xc4) >> 4)] =
                (unsigned short)m16;
        }
        // ---- stage W1 tile col-major (fp32 -> bf16) ---------------------
        {
            unsigned short tmpw[64] __attribute__((aligned(16)));
#pragma unroll
            for (int k = 0; k < 64; ++k)
                tmpw[k] = f2bf(W1[(size_t)(kk + k) * MM + w1c]);
            int4* dst = (int4*)&w1T[w1c][0];
#pragma unroll
            for (int i = 0; i < 8; ++i) dst[i] = ((const int4*)tmpw)[i];
        }
        __syncthreads();

        // ---- 2 WMMA k-steps of 32 ---------------------------------------
#pragma unroll
        for (int ks = 0; ks < 2; ++ks) {
            FragU A;
            const int arow = rowW + (lane & 15);
            const int akb  = ks * 32 + ((lane >> 4) << 3);
            A.q[0] = *(const int4*)&xT[arow][akb];
            A.q[1] = *(const int4*)&xT[arow][akb + 16];
#pragma unroll
            for (int t = 0; t < 8; ++t) {
                FragU Bm;
                const int bcol = colW + t * 16 + (lane & 15);
                const int bkb  = ks * 32 + ((lane >> 4) << 3);
                const int4* bp = (const int4*)&w1T[bcol][bkb];
                Bm.q[0] = bp[0];
                Bm.q[1] = bp[2];
                acc[t] = __builtin_amdgcn_wmma_f32_16x16x32_bf16(
                    false, A.v, false, Bm.v, (short)0, acc[t], false, false);
            }
        }
        __syncthreads();
    }

    // ---- epilogue: +b1, relu, bf16 store to H ---------------------------
#pragma unroll
    for (int t = 0; t < 8; ++t) {
        const int c = colW + t * 16 + (lane & 15);
        const float bias = b1s[c];
#pragma unroll
        for (int i = 0; i < 8; ++i) {
            const int r = rowW + i + ((lane >> 4) << 3);
            float f = acc[t][i] + bias;
            f = fmaxf(f, 0.0f);
            Hout[(size_t)(rowBase + r) * MM + c] = f2bf(f);
        }
    }
}

// ---------------------------------------------------------------------------
// Kernel 2: f = H @ W2 + b2 -> d_out; masked row-sums via LDS ds_add_f32;
// then in-place y = mask ? f/s_safe : 0 for this block's 64 rows.
// grid = B/64 blocks, 256 threads (8 waves; wave grid 4 rows x 2 cols;
// each wave owns 16 rows x 32 cols = 2 WMMA tiles per 64-wide N chunk).
// ---------------------------------------------------------------------------
__launch_bounds__(256)
__global__ void k2_gemm2_norm(const unsigned short* __restrict__ Hbuf,
                              const float* __restrict__ W2,
                              const float* __restrict__ b2,
                              const unsigned short* __restrict__ maskBuf,
                              float* __restrict__ out) {
    __shared__ unsigned short w2T[64][264];  // [col][k] col stride 528B (16B aligned)
    __shared__ float b2s[64];
    __shared__ unsigned short maskLds[64 * 4];
    __shared__ float sRow[64];

    const int tid  = threadIdx.x;
    const int lane = tid & 31;
    const int wave = tid >> 5;
    const int waveRow = wave >> 1;
    const int waveCol = wave & 1;
    const int rowW = waveRow << 4;
    const int colW = waveCol << 5;
    const int rowBase = blockIdx.x * 64;

    if (tid < 64) sRow[tid] = 0.0f;
    __syncthreads();

    const int w2c = tid & 63;                // column within chunk
    const int w2q = tid >> 6;                // k quarter (0..3) -> 64 k each

    for (int nb = 0; nb < NN; nb += 64) {
        // ---- stage W2 chunk col-major bf16 ------------------------------
        {
            unsigned short tmpw[64] __attribute__((aligned(16)));
#pragma unroll
            for (int k = 0; k < 64; ++k)
                tmpw[k] = f2bf(W2[(size_t)(w2q * 64 + k) * NN + nb + w2c]);
            int4* dst = (int4*)&w2T[w2c][w2q * 64];
#pragma unroll
            for (int i = 0; i < 8; ++i) dst[i] = ((const int4*)tmpw)[i];
        }
        if (tid < 64) b2s[tid] = b2[nb + tid];
        {
            const int r = tid >> 2, w = tid & 3;
            maskLds[tid] = maskBuf[(size_t)(rowBase + r) * (NN / 16) + (nb >> 4) + w];
        }
        if (nb + 64 < NN)  // pull next W2 chunk toward L2 (global_prefetch_b8)
            __builtin_prefetch(&W2[(size_t)(w2q * 64) * NN + nb + 64 + w2c], 0, 1);
        __syncthreads();

        v8f acc[2];
        acc[0] = (v8f)0.0f;
        acc[1] = (v8f)0.0f;

#pragma unroll
        for (int ks = 0; ks < 8; ++ks) {
            FragU A;
            const int arow = rowBase + rowW + (lane & 15);
            const int akb  = ks * 32 + ((lane >> 4) << 3);
            const int4* hp = (const int4*)(Hbuf + (size_t)arow * MM + akb);
            A.q[0] = hp[0];
            A.q[1] = hp[2];  // +16 bf16 = +32B
#pragma unroll
            for (int t = 0; t < 2; ++t) {
                FragU Bm;
                const int bcol = colW + t * 16 + (lane & 15);
                const int bkb  = ks * 32 + ((lane >> 4) << 3);
                const int4* bp = (const int4*)&w2T[bcol][bkb];
                Bm.q[0] = bp[0];
                Bm.q[1] = bp[2];
                acc[t] = __builtin_amdgcn_wmma_f32_16x16x32_bf16(
                    false, A.v, false, Bm.v, (short)0, acc[t], false, false);
            }
        }

        // ---- epilogue: +b2, store f, masked row-sum ---------------------
#pragma unroll
        for (int t = 0; t < 2; ++t) {
            const int cL = colW + t * 16 + (lane & 15);
            const float bias = b2s[cL];
#pragma unroll
            for (int i = 0; i < 8; ++i) {
                const int rL = rowW + i + ((lane >> 4) << 3);
                const float f = acc[t][i] + bias;
                out[(size_t)(rowBase + rL) * NN + nb + cL] = f;
                const unsigned m = maskLds[rL * 4 + (cL >> 4)];
                if ((m >> (cL & 15)) & 1u) atomicAdd(&sRow[rL], f);
            }
        }
        __syncthreads();
    }

    __threadfence();   // make this block's f stores visible to its own reads
    __syncthreads();

    // ---- in-place normalize: y = mask ? f / s_safe : 0 ------------------
    for (int iter = 0; iter < 256; ++iter) {
        const int idx = iter * 256 + tid;        // float4 index in block region
        const int r   = idx >> 10;               // 1024 float4 per row
        const int c4  = (idx & 1023) << 2;
        float s = sRow[r];
        s = (s == 0.0f) ? 1.0f : s;
        const float rs = 1.0f / s;
        const size_t off = (size_t)(rowBase + r) * NN + c4;
        float4 f = *(const float4*)(out + off);
        const unsigned m =
            maskBuf[(size_t)(rowBase + r) * (NN / 16) + (c4 >> 4)];
        const int bb = c4 & 15;
        f.x = ((m >> (bb + 0)) & 1u) ? f.x * rs : 0.0f;
        f.y = ((m >> (bb + 1)) & 1u) ? f.y * rs : 0.0f;
        f.z = ((m >> (bb + 2)) & 1u) ? f.z * rs : 0.0f;
        f.w = ((m >> (bb + 3)) & 1u) ? f.w * rs : 0.0f;
        *(float4*)(out + off) = f;
    }
}

// ---------------------------------------------------------------------------
extern "C" void kernel_launch(void* const* d_in, const int* in_sizes, int n_in,
                              void* d_out, int out_size, void* d_ws, size_t ws_size,
                              hipStream_t stream) {
    const float* x  = (const float*)d_in[0];
    const float* W1 = (const float*)d_in[1];
    const float* b1 = (const float*)d_in[2];
    const float* W2 = (const float*)d_in[3];
    const float* b2 = (const float*)d_in[4];
    float* out = (float*)d_out;

    // ws layout: H bf16 [B,256] (4 MB) | mask u16 [B, N/16] (4 MB)
    unsigned short* Hbuf    = (unsigned short*)d_ws;
    unsigned short* maskBuf = Hbuf + (size_t)BB * MM;

    k1_gemm1_relu_mask<<<BB / 64, 256, 0, stream>>>(x, W1, b1, Hbuf, maskBuf);
    k2_gemm2_norm<<<BB / 64, 256, 0, stream>>>(Hbuf, W2, b2, maskBuf, out);
}
